// GraphRelationUpdate_53884659695843
// MI455X (gfx1250) — matrix operand order repaired and checked by reading, not compiled, not run on hardware
//
#include <hip/hip_runtime.h>
#include <hip/hip_bf16.h>
#include <math.h>

// ---------------------------------------------------------------------------
// GraphRelationUpdate for MI455X (gfx1250, wave32)
//   Kernel 1: node_score = (LeakyReLU(e @ Wa^T + b)) @ ua   via v_wmma_f32_16x16x32_f16
//   Kernel 2: out = e + softmax_k(score[nbr]) * e[nbr]      (L2-resident gathers)
// ---------------------------------------------------------------------------

typedef __attribute__((ext_vector_type(16))) _Float16 v16h;
typedef __attribute__((ext_vector_type(8)))  _Float16 v8h;
typedef __attribute__((ext_vector_type(8)))  float    v8f;
typedef __attribute__((ext_vector_type(4)))  float    v4f;

#define HDIM 128
#define KNBR 16
#define LDS_STRIDE 136   // 128 + 8 halves pad -> 272B row stride -> conflict-free B reads

// ---------------------------------------------------------------------------
// Kernel 1: per-node attention score.
// Block = 256 threads = 8 waves; each wave computes a 16-row tile of h and
// reduces it against ua.  32 WMMAs per wave (8 column tiles x 4 k-steps).
// ---------------------------------------------------------------------------
__global__ __launch_bounds__(256)
void grub_score_kernel(const float* __restrict__ e,
                       const float* __restrict__ Wa,      // [H,H] row-major [out,in]
                       const float* __restrict__ Wb,      // [H]
                       const float* __restrict__ ua,      // [H]
                       float* __restrict__ score,         // [N]
                       int numTiles, int N)
{
    __shared__ _Float16 w_lds[HDIM * LDS_STRIDE];

    const int tid = threadIdx.x;

    // Stage Wa -> LDS as f16 (whole block cooperates; 64 elements per thread).
    #pragma unroll
    for (int i = 0; i < (HDIM * HDIM) / 256; ++i) {
        const int idx = tid + i * 256;
        const int n = idx >> 7;          // output channel (row of Wa)
        const int k = idx & (HDIM - 1);  // input channel
        w_lds[n * LDS_STRIDE + k] = (_Float16)Wa[idx];
    }
    __syncthreads();

    const int wave = tid >> 5;
    const int lane = tid & 31;
    const int tile = blockIdx.x * 8 + wave;
    if (tile >= numTiles) return;        // wave-uniform: EXEC stays all-1 below

    const int m0       = tile * 16;
    const int lrow     = lane & 15;      // column-within-tile for B/C, row for A
    const int half_sel = lane >> 4;      // which 16-lane half
    const int koff     = half_sel * 8;   // A-fragment K interleave (ISA table)
    int arow = m0 + lrow;
    if (arow > N - 1) arow = N - 1;      // address clamp only (N%16==0 normally)

    const float* erow = e + (long)arow * HDIM;

    v8f acc[8] = {};                     // 8 column tiles of C (f32)

    #pragma unroll
    for (int kk = 0; kk < HDIM; kk += 32) {
        // ---- A fragment: lane holds row `arow`, K = kk+koff+{0..7, 16..23}
        v4f a0 = *(const v4f*)(erow + kk + koff);
        v4f a1 = *(const v4f*)(erow + kk + koff + 4);
        v4f a2 = *(const v4f*)(erow + kk + koff + 16);
        v4f a3 = *(const v4f*)(erow + kk + koff + 20);
        v16h a;
        #pragma unroll
        for (int j = 0; j < 4; ++j) {
            a[j]      = (_Float16)a0[j];
            a[4 + j]  = (_Float16)a1[j];
            a[8 + j]  = (_Float16)a2[j];
            a[12 + j] = (_Float16)a3[j];
        }
        // ---- 8 column tiles: B[k, n] = Wa[n, k]; lane half h holds K = 8h+{0..7, 16..23}
        #pragma unroll
        for (int t = 0; t < 8; ++t) {
            const int n = t * 16 + lrow;
            const _Float16* bp = &w_lds[n * LDS_STRIDE + kk + half_sel * 8];
            v8h b0 = *(const v8h*)bp;
            v8h b1 = *(const v8h*)(bp + 16);
            v16h b;
            #pragma unroll
            for (int j = 0; j < 8; ++j) { b[j] = b0[j]; b[8 + j] = b1[j]; }
            acc[t] = __builtin_amdgcn_wmma_f32_16x16x32_f16(
                         false, a, false, b, (short)0, acc[t], false, false);
        }
    }

    // Epilogue: bias + LeakyReLU(0.1), dot with ua, reduce columns -> rows.
    float partial[8];
    #pragma unroll
    for (int r = 0; r < 8; ++r) partial[r] = 0.f;

    #pragma unroll
    for (int t = 0; t < 8; ++t) {
        const int n = t * 16 + lrow;
        const float bias = Wb[n];
        const float u    = ua[n];
        #pragma unroll
        for (int r = 0; r < 8; ++r) {
            float v = acc[t][r] + bias;
            float h = (v > 0.f) ? v : 0.1f * v;
            partial[r] += h * u;
        }
    }
    // Sum over the 16 columns held by each 16-lane group.
    #pragma unroll
    for (int off = 1; off < 16; off <<= 1) {
        #pragma unroll
        for (int r = 0; r < 8; ++r)
            partial[r] += __shfl_xor(partial[r], off, 16);
    }
    // Lane 0 of each half writes its 8 rows (half 0 -> rows 0..7, half 1 -> 8..15).
    if (lrow == 0) {
        #pragma unroll
        for (int r = 0; r < 8; ++r) {
            const int row = m0 + half_sel * 8 + r;
            if (row < N) score[row] = partial[r];
        }
    }
}

// ---------------------------------------------------------------------------
// Kernel 2: softmax over 16 neighbor scores + weighted aggregation + residual.
// One wave32 per node; each lane owns a float4 slice of the 128-wide row, so a
// neighbor-row gather is one fully coalesced 512B wave access (L2-resident).
// ---------------------------------------------------------------------------
__global__ __launch_bounds__(256)
void grub_agg_kernel(const float* __restrict__ e,
                     const int*   __restrict__ nbr,    // [N,16]
                     const float* __restrict__ score,  // [N]
                     float* __restrict__ out,          // [N,H]
                     int N)
{
    const int wave = threadIdx.x >> 5;
    const int lane = threadIdx.x & 31;
    const int node = blockIdx.x * 8 + wave;
    if (node >= N) return;

    const int k16 = lane & 15;          // both halves mirror the 16 neighbors
    const int idx = nbr[node * KNBR + k16];
    float s = score[idx];

    // max over the 16-lane group
    float m = s;
    #pragma unroll
    for (int off = 1; off < 16; off <<= 1)
        m = fmaxf(m, __shfl_xor(m, off, 16));
    const float ex = __expf(s - m);
    float sum = ex;
    #pragma unroll
    for (int off = 1; off < 16; off <<= 1)
        sum += __shfl_xor(sum, off, 16);
    const float w = ex / sum;

    // Residual init, then accumulate the 16 weighted neighbor rows.
    const v4f* __restrict__ self = (const v4f*)(e + (long)node * HDIM);
    v4f acc = self[lane];
    #pragma unroll
    for (int k = 0; k < KNBR; ++k) {
        const float wk = __shfl(w,   k, 16);
        const int   nk = __shfl(idx, k, 16);
        const v4f v = ((const v4f*)(e + (long)nk * HDIM))[lane];
        acc += wk * v;
    }
    ((v4f*)(out + (long)node * HDIM))[lane] = acc;
}

// ---------------------------------------------------------------------------
extern "C" void kernel_launch(void* const* d_in, const int* in_sizes, int n_in,
                              void* d_out, int out_size, void* d_ws, size_t ws_size,
                              hipStream_t stream)
{
    const float* e    = (const float*)d_in[0];   // [N,128]
    const float* Wa_w = (const float*)d_in[1];   // [128,128]
    const float* Wa_b = (const float*)d_in[2];   // [128]
    const float* ua   = (const float*)d_in[3];   // [128]
    const int*   nbr  = (const int*)d_in[4];     // [N,16]
    float*       out  = (float*)d_out;           // [N,128]
    float*       scr  = (float*)d_ws;            // [N] node scores

    const int N = in_sizes[0] / HDIM;
    const int numTiles = (N + 15) / 16;

    dim3 block(256);
    grub_score_kernel<<<dim3((numTiles + 7) / 8), block, 0, stream>>>(
        e, Wa_w, Wa_b, ua, scr, numTiles, N);
    grub_agg_kernel<<<dim3((N + 7) / 8), block, 0, stream>>>(
        e, nbr, scr, out, N);
}
